// SNNLanguageModel_55576876810900
// MI455X (gfx1250) — matrix operand order repaired and checked by reading, not compiled, not run on hardware
//
#include <hip/hip_runtime.h>
#include <hip/hip_bf16.h>

#define T_ 512
#define B_ 16
#define H_ 256
#define V_ 32000
#define L_ 3

#define SCAN_C 8              // time chunks for parallel scan
#define SCAN_S (T_ / SCAN_C)  // 64 steps per chunk

typedef __attribute__((ext_vector_type(16))) __bf16 v16bf;
typedef __attribute__((ext_vector_type(8)))  __bf16 bf16x8;
typedef __attribute__((ext_vector_type(8)))  float  v8f;

__device__ inline __bf16 f2bf(float f) {
  unsigned u = __builtin_bit_cast(unsigned, f);
  u += 0x7FFFu + ((u >> 16) & 1u);          // round-to-nearest-even
  return __builtin_bit_cast(__bf16, (unsigned short)(u >> 16));
}
__device__ inline float bf2f(__bf16 x) {
  unsigned u = ((unsigned)__builtin_bit_cast(unsigned short, x)) << 16;
  return __builtin_bit_cast(float, u);
}

// ---------------------------------------------------------------- fp32 -> bf16
__global__ __launch_bounds__(256) void f32_to_bf16_kernel(
    const float* __restrict__ src, __bf16* __restrict__ dst, int n) {
  int i = blockIdx.x * blockDim.x + threadIdx.x;
  if (i < n) dst[i] = f2bf(src[i]);
}

// --------------------------------------------------------------- scan input
template <bool EMBED>
__device__ inline float load_x(int t, int b, int hc,
    const int* __restrict__ tokens, const float* __restrict__ tok_emb,
    const float* __restrict__ pos_emb, const __bf16* __restrict__ xin) {
  if (EMBED) {
    const int tok = tokens[t * B_ + b];
    return tok_emb[tok * H_ + hc] + pos_emb[t * H_ + hc];
  } else {
    return bf2f(xin[(t * B_ + b) * H_ + hc]);
  }
}

// ------------------------------------------- phase 1: per-chunk carry (v from 0)
// thread gid = c*(B_*H_) + b*H_ + hc ; 8*4096 = 32768 threads = 1024 waves
template <bool EMBED>
__global__ __launch_bounds__(256) void scan_carry_kernel(
    const int* __restrict__ tokens, const float* __restrict__ tok_emb,
    const float* __restrict__ pos_emb, const __bf16* __restrict__ xin,
    const float* __restrict__ beta, float* __restrict__ carry) {
  const int gid = blockIdx.x * blockDim.x + threadIdx.x;
  const int c   = gid >> 12;          // / (B_*H_)
  const int idx = gid & 4095;
  const int b   = idx >> 8;
  const int hc  = idx & (H_ - 1);
  const float bet = beta[hc];
  const int t0 = c * SCAN_S;
  float v = 0.f;
#pragma unroll 8
  for (int s = 0; s < SCAN_S; ++s)
    v = fmaf(bet, v, load_x<EMBED>(t0 + s, b, hc, tokens, tok_emb, pos_emb, xin));
  carry[gid] = v;
}

// -------- phase 2: rebuild chunk-entry state from carries, replay local scan
template <bool EMBED>
__global__ __launch_bounds__(256) void scan_apply_kernel(
    const int* __restrict__ tokens, const float* __restrict__ tok_emb,
    const float* __restrict__ pos_emb, const __bf16* __restrict__ xin,
    const float* __restrict__ beta, const float* __restrict__ carry,
    __bf16* __restrict__ vout) {
  const int gid = blockIdx.x * blockDim.x + threadIdx.x;
  const int c   = gid >> 12;
  const int idx = gid & 4095;
  const int b   = idx >> 8;
  const int hc  = idx & (H_ - 1);
  const float bet = beta[hc];

  float b64 = bet;                                   // beta^64 via 6 squarings
#pragma unroll
  for (int i = 0; i < 6; ++i) b64 *= b64;

  float v = 0.f;                                     // P_c: state entering chunk c
  for (int cc = 0; cc < c; ++cc)
    v = fmaf(b64, v, carry[cc * (B_ * H_) + idx]);

  const int t0 = c * SCAN_S;
#pragma unroll 8
  for (int s = 0; s < SCAN_S; ++s) {
    v = fmaf(bet, v, load_x<EMBED>(t0 + s, b, hc, tokens, tok_emb, pos_emb, xin));
    vout[((t0 + s) * B_ + b) * H_ + hc] = f2bf(v);
  }
}

// ------------------------------------------------------------- WMMA GEMM
// out[m,n] = act( sum_k A[m,k]*W[n,k] + bias[n] )
// A: 8192 x 256 bf16 row-major; W: N x 256 bf16 row-major.
// Wave computes a 16x64 strip (4 accumulators); 8 waves/block share one
// 64-row W strip staged in LDS (32 KB).
template <bool RELU, bool OUT_BF16>
__global__ __launch_bounds__(256) void gemm_wmma_kernel(
    const __bf16* __restrict__ A, const __bf16* __restrict__ W,
    const float* __restrict__ bias, void* __restrict__ outv, int N) {
  __shared__ __bf16 Wlds[64 * H_];                 // 32768 bytes

  const int lane   = threadIdx.x & 31;
  const int waveId = blockIdx.x * 8 + (threadIdx.x >> 5);
  const int mt     = waveId & 511;                 // M/16 = 512 tiles
  const int strip  = waveId >> 9;                  // N/64 strip (same in block)
  const int Mbase  = mt << 4;
  const int Nbase  = strip << 6;

  // stage W[Nbase .. Nbase+63][0..255] (contiguous 16384 bf16) into LDS
  {
    const bf16x8* g = (const bf16x8*)(W + Nbase * H_);
    bf16x8* l = (bf16x8*)Wlds;
#pragma unroll
    for (int i = 0; i < 8; ++i) {
      const int e = i * 256 + threadIdx.x;         // 2048 16-byte chunks
      l[e] = g[e];
    }
  }
  __syncthreads();

  const int half = lane >> 4;                      // K half select
  const int lr   = lane & 15;                      // tile row / N lane

  v8f acc0 = {}, acc1 = {}, acc2 = {}, acc3 = {};
  const __bf16* Arow = A + (Mbase + lr) * H_;
#pragma unroll
  for (int kk = 0; kk < H_; kk += 32) {
    // A lane layout: K {kk..kk+15} (half 0); +16 for half 1
    const int k0 = kk + half * 16;
    bf16x8 alo = *(const bf16x8*)(Arow + k0);
    bf16x8 ahi = *(const bf16x8*)(Arow + k0 + 8);
    v16bf a = __builtin_shufflevector(alo, ahi,
        0,1,2,3,4,5,6,7,8,9,10,11,12,13,14,15);
    // load ALL four B tiles first (distinct registers) so the scheduler can
    // clause the 8 ds_loads, wait once, then issue 4 back-to-back WMMAs
    const int kb = kk + half * 16;                 // B: contiguous 16-wide K run
    v16bf b0 = *(const v16bf*)(Wlds + (0 * 16 + lr) * H_ + kb);
    v16bf b1 = *(const v16bf*)(Wlds + (1 * 16 + lr) * H_ + kb);
    v16bf b2 = *(const v16bf*)(Wlds + (2 * 16 + lr) * H_ + kb);
    v16bf b3 = *(const v16bf*)(Wlds + (3 * 16 + lr) * H_ + kb);
    acc0 = __builtin_amdgcn_wmma_f32_16x16x32_bf16(false, a, false, b0, (short)0, acc0, false, false);
    acc1 = __builtin_amdgcn_wmma_f32_16x16x32_bf16(false, a, false, b1, (short)0, acc1, false, false);
    acc2 = __builtin_amdgcn_wmma_f32_16x16x32_bf16(false, a, false, b2, (short)0, acc2, false, false);
    acc3 = __builtin_amdgcn_wmma_f32_16x16x32_bf16(false, a, false, b3, (short)0, acc3, false, false);
  }

  v8f accs[4] = {acc0, acc1, acc2, acc3};
#pragma unroll
  for (int j = 0; j < 4; ++j) {
    const int n  = Nbase + j * 16 + lr;
    const float bn = bias[n];
#pragma unroll
    for (int r = 0; r < 8; ++r) {
      const int m = Mbase + r + half * 8;          // C/D VGPR layout
      float v = accs[j][r] + bn;
      if (RELU) v = v > 0.f ? v : 0.f;
      if (OUT_BF16) ((__bf16*)outv)[m * N + n] = f2bf(v);
      else          ((float*)outv)[m * N + n] = v;
    }
  }
}

// ------------------------------------------------------------------- launch
extern "C" void kernel_launch(void* const* d_in, const int* in_sizes, int n_in,
                              void* d_out, int out_size, void* d_ws, size_t ws_size,
                              hipStream_t stream) {
  const int*   tokens  = (const int*)d_in[0];
  const float* tok_emb = (const float*)d_in[1];
  const float* pos_emb = (const float*)d_in[2];
  const float* betas   = (const float*)d_in[3];
  const float* fc_w    = (const float*)d_in[4];
  const float* fc_b    = (const float*)d_in[5];
  const float* out_w   = (const float*)d_in[6];
  const float* out_b   = (const float*)d_in[7];
  float* out = (float*)d_out;

  char* ws = (char*)d_ws;
  __bf16* vbuf = (__bf16*)(ws);                    //  4 MiB (T*B*H bf16)
  __bf16* hbuf = (__bf16*)(ws + (4u  << 20));      //  4 MiB
  __bf16* fcw  = (__bf16*)(ws + (8u  << 20));      //  384 KiB (L*H*H bf16)
  __bf16* outw = (__bf16*)(ws + (9u  << 20));      //  ~15.6 MiB (V*H bf16)
  float*  cbuf = (float*)(ws + (25u << 20));       //  128 KiB (SCAN_C*B*H f32)

  // weight precision conversion (bf16 WMMA path)
  f32_to_bf16_kernel<<<(L_ * H_ * H_ + 255) / 256, 256, 0, stream>>>(fc_w, fcw, L_ * H_ * H_);
  f32_to_bf16_kernel<<<(V_ * H_ + 255) / 256, 256, 0, stream>>>(out_w, outw, V_ * H_);

  const int scanBlocks = (SCAN_C * B_ * H_) / 256;  // 128 blocks = 1024 waves
  for (int l = 0; l < L_; ++l) {
    const float* beta = betas + l * H_;
    if (l == 0) {
      scan_carry_kernel<true><<<scanBlocks, 256, 0, stream>>>(
          tokens, tok_emb, pos_emb, nullptr, beta, cbuf);
      scan_apply_kernel<true><<<scanBlocks, 256, 0, stream>>>(
          tokens, tok_emb, pos_emb, nullptr, beta, cbuf, vbuf);
    } else {
      scan_carry_kernel<false><<<scanBlocks, 256, 0, stream>>>(
          nullptr, nullptr, nullptr, hbuf, beta, cbuf);
      scan_apply_kernel<false><<<scanBlocks, 256, 0, stream>>>(
          nullptr, nullptr, nullptr, hbuf, beta, cbuf, vbuf);
    }
    // M=8192 -> 512 tiles, N=256 -> 4 strips => 2048 waves / 8 = 256 blocks
    gemm_wmma_kernel<true, true><<<256, 256, 0, stream>>>(
        vbuf, fcw + l * H_ * H_, fc_b + l * H_, (void*)hbuf, H_);
  }

  // logits: M=8192 -> 512 tiles, N=32000 -> 500 strips => 256000 waves / 8
  gemm_wmma_kernel<false, false><<<32000, 256, 0, stream>>>(
      hbuf, outw, out_b, (void*)out, V_);
}
